// GroupedQueryAttention_27212912787981
// MI455X (gfx1250) — compile-verified, matrix operand order checked by reading, not run on hardware
//
#include <hip/hip_runtime.h>

// ---------------------------------------------------------------- constants
#define B_    2
#define S_    2048
#define D_    2048
#define HQ_   32
#define HKV_  8
#define HD_   64
#define SCALE_ 0.125f            // 1/sqrt(64)

typedef __attribute__((ext_vector_type(16))) __bf16 v16bf;
typedef __attribute__((ext_vector_type(8)))  __bf16 v8bf;
typedef __attribute__((ext_vector_type(8)))  float  v8f;
typedef __attribute__((ext_vector_type(4)))  int    v4i;

typedef __attribute__((address_space(1))) v4i gv4i;   // global int4 (async src)
typedef __attribute__((address_space(3))) v4i lv4i;   // LDS int4 (async dst)

// ------------------------------------------------------------ bf16 helpers
static __device__ __forceinline__ __bf16 f2bf(float f) {
  unsigned int u = __builtin_bit_cast(unsigned int, f);
  unsigned int r = (u + 0x7fffu + ((u >> 16) & 1u)) >> 16;   // RNE
  unsigned short s = (unsigned short)r;
  return __builtin_bit_cast(__bf16, s);
}
static __device__ __forceinline__ float bf2f(__bf16 b) {
  unsigned short s = __builtin_bit_cast(unsigned short, b);
  unsigned int u = ((unsigned int)s) << 16;
  return __builtin_bit_cast(float, u);
}
static __device__ __forceinline__ v16bf cat16(v8bf lo, v8bf hi) {
  v16bf r;
#pragma unroll
  for (int i = 0; i < 8; ++i) { r[i] = lo[i]; r[8 + i] = hi[i]; }
  return r;
}

static __device__ __forceinline__ float loadA(const float* p, int i)  { return p[i]; }
static __device__ __forceinline__ float loadA(const __bf16* p, int i) { return bf2f(p[i]); }
static __device__ __forceinline__ void  storeC(float* p, size_t i, float v)  { p[i] = v; }
static __device__ __forceinline__ void  storeC(__bf16* p, size_t i, float v) { p[i] = f2bf(v); }

// ------------------------------------------------------------------- GEMM
// C[M,N] = A[M,K] * W[K,N]; bf16 WMMA multiplies, fp32 accumulate.
// Block: 256 threads = 8 waves in a 4(M)x2(N) grid.
// Block tile 128x128, K-step 32; wave tile 32x64 -> 8 WMMAs per K-step/wave.
// When A is already bf16 the A tile is staged with async LDS loads
// (GLOBAL_LOAD_ASYNC_TO_LDS_B128 / ASYNCcnt) when the toolchain exposes it.
template <typename TA, typename TC>
__global__ __launch_bounds__(256) void gemm_wmma(const TA* __restrict__ A,
                                                 const float* __restrict__ W,
                                                 TC* __restrict__ C,
                                                 int M, int N, int K) {
  __shared__ __bf16 As[128][32];   // [m][k]
  __shared__ __bf16 Bs[128][32];   // [n][k]  (W staged transposed)

  const int t    = threadIdx.x;
  const int lane = t & 31;
  const int wave = t >> 5;
  const int wm   = wave >> 1;     // 0..3 -> 32-row slice
  const int wn   = wave & 1;      // 0..1 -> 64-col slice
  const int m0   = blockIdx.y * 128;
  const int n0   = blockIdx.x * 128;
  const int r16  = lane & 15;
  const int kh   = (lane >> 4) * 8;

  v8f acc[2][4] = {};

  const int ar = t >> 1;              // A stage: row 0..127
  const int ac = (t & 1) * 16;        //          k 0 or 16, 16 elems
  const int bk = t >> 3;              // B stage: k 0..31
  const int bc = (t & 7) * 16;        //          n offset, 16 elems

  for (int k0 = 0; k0 < K; k0 += 32) {
    bool used_async = false;
#if defined(__gfx1250__) && __has_builtin(__builtin_amdgcn_global_load_async_to_lds_b128)
    if constexpr (__is_same(TA, __bf16)) {
      // A tile is 128x32 bf16 = 8 KB = 512 x 16B chunks; 2 chunks per thread.
      // Pure byte copy -> async DMA to LDS, no VGPR round-trip.
#pragma unroll
      for (int i = 0; i < 2; ++i) {
        int byte = (t * 2 + i) * 16;
        int row  = byte >> 6;          // 64 B per As row
        int col  = (byte & 63) >> 1;   // element offset within row
        const TA* g = A + (size_t)(m0 + row) * K + k0 + col;
        __builtin_amdgcn_global_load_async_to_lds_b128(
            (gv4i*)const_cast<TA*>(g), (lv4i*)&As[row][col], 0, 0);
      }
      used_async = true;
    }
#endif
    if (!used_async) { // stage A tile 128x32 with fp32->bf16 conversion
      const TA* src = A + (size_t)(m0 + ar) * K + k0 + ac;
#pragma unroll
      for (int i = 0; i < 16; ++i) As[ar][ac + i] = f2bf(loadA(src, i));
    }
    { // stage B tile (W is K-major): Bs[n][k]
      const float* src = W + (size_t)(k0 + bk) * N + n0 + bc;
#pragma unroll
      for (int i = 0; i < 16; ++i) Bs[bc + i][bk] = f2bf(src[i]);
    }
    // prefetch next K-step tiles into cache (global_prefetch_b8);
    // 192 MB L2 holds the whole working set.
    if (k0 + 32 < K) {
      __builtin_prefetch(A + (size_t)(m0 + ar) * K + k0 + 32 + ac, 0, 3);
      __builtin_prefetch(W + (size_t)(k0 + 32 + bk) * N + n0 + bc, 0, 3);
    }
#if defined(__gfx1250__) && __has_builtin(__builtin_amdgcn_global_load_async_to_lds_b128)
    if (used_async) {
#if __has_builtin(__builtin_amdgcn_s_wait_asynccnt)
      __builtin_amdgcn_s_wait_asynccnt(0);
#else
      asm volatile("s_wait_asynccnt 0" ::: "memory");
#endif
    }
#endif
    __syncthreads();

    v16bf a0, a1, bfr[4];
    {
      const int row = wm * 32 + r16;
      a0 = cat16(*(const v8bf*)&As[row][kh],      *(const v8bf*)&As[row][16 + kh]);
      a1 = cat16(*(const v8bf*)&As[row + 16][kh], *(const v8bf*)&As[row + 16][16 + kh]);
    }
#pragma unroll
    for (int j = 0; j < 4; ++j) {
      const int nn = wn * 64 + j * 16 + r16;
      bfr[j] = cat16(*(const v8bf*)&Bs[nn][kh], *(const v8bf*)&Bs[nn][16 + kh]);
    }
#if defined(__gfx1250__)
#pragma unroll
    for (int j = 0; j < 4; ++j) {
      acc[0][j] = __builtin_amdgcn_wmma_f32_16x16x32_bf16(false, a0, false, bfr[j], (short)0, acc[0][j], false, false);
      acc[1][j] = __builtin_amdgcn_wmma_f32_16x16x32_bf16(false, a1, false, bfr[j], (short)0, acc[1][j], false, false);
    }
#endif
    __syncthreads();
  }

  const int half = lane >> 4;
#pragma unroll
  for (int i = 0; i < 2; ++i) {
#pragma unroll
    for (int v = 0; v < 8; ++v) {
      size_t rm = (size_t)(m0 + wm * 32 + i * 16 + v + 8 * half) * N;
#pragma unroll
      for (int j = 0; j < 4; ++j)
        storeC(C, rm + n0 + wn * 64 + j * 16 + r16, acc[i][j][v]);
    }
  }
}

// ------------------------------------------------------------------- RoPE
// layout [b][s][h][d], d in [0,64)
__global__ void rope_kernel(const __bf16* __restrict__ in,
                            const float* __restrict__ cosb,
                            const float* __restrict__ sinb,
                            __bf16* __restrict__ out, int H, int total) {
  int idx = blockIdx.x * blockDim.x + threadIdx.x;
  if (idx >= total) return;
  int d = idx & 63;
  int s = (idx / (64 * H)) % S_;
  float v = bf2f(in[idx]);
  float c = cosb[s * 64 + d];
  float sn = sinb[s * 64 + d];
  float other = (d < 32) ? -bf2f(in[idx + 32]) : bf2f(in[idx - 32]);
  out[idx] = f2bf(v * c + other * sn);
}

// --------------------------------------------------------- flash attention
// grid: (S/64, B*HQ); block 128 = 4 waves; wave owns 16 query rows.
__global__ __launch_bounds__(128) void flash_kernel(const __bf16* __restrict__ Q,
                                                    const __bf16* __restrict__ Kc,
                                                    const __bf16* __restrict__ V,
                                                    __bf16* __restrict__ O) {
  __shared__ __bf16 Ps[4][16][32];   // per-wave P tile (q x key)
  __shared__ __bf16 Vs[4][64][32];   // per-wave V^T tile (d x key)

  const int lane = threadIdx.x & 31;
  const int wave = threadIdx.x >> 5;
  const int bh = blockIdx.y;
  const int b  = bh / HQ_;
  const int h  = bh % HQ_;
  const int hk = h >> 2;                       // N_REP = 4
  const int qbase = blockIdx.x * 64 + wave * 16;
  const int q_hi  = qbase + 15;                // wave-uniform causal bound
  const int r16   = lane & 15;
  const int kh    = (lane >> 4) * 8;
  const int half  = lane >> 4;

  // Q fragments for d=[0,32) and [32,64)
  const __bf16* qrow = Q + (((size_t)b * S_ + (qbase + r16)) * HQ_ + h) * HD_;
  v16bf qf0 = cat16(*(const v8bf*)(qrow + kh),      *(const v8bf*)(qrow + 16 + kh));
  v16bf qf1 = cat16(*(const v8bf*)(qrow + 32 + kh), *(const v8bf*)(qrow + 48 + kh));

  float m_i[8], l_i[8];
  v8f acc[4] = {};
#pragma unroll
  for (int v = 0; v < 8; ++v) { m_i[v] = -1e30f; l_i[v] = 0.f; }

  for (int kb = 0; kb <= q_hi; kb += 32) {
    // ---- scores: S = Q K^T for two 16-key tiles
    v8f s0 = {}, s1 = {};
    {
      const __bf16* kr0 = Kc + (((size_t)b * S_ + (kb + r16)) * HKV_ + hk) * HD_;
      const __bf16* kr1 = Kc + (((size_t)b * S_ + (kb + 16 + r16)) * HKV_ + hk) * HD_;
      v16bf k00 = cat16(*(const v8bf*)(kr0 + kh),      *(const v8bf*)(kr0 + 16 + kh));
      v16bf k01 = cat16(*(const v8bf*)(kr0 + 32 + kh), *(const v8bf*)(kr0 + 48 + kh));
      v16bf k10 = cat16(*(const v8bf*)(kr1 + kh),      *(const v8bf*)(kr1 + 16 + kh));
      v16bf k11 = cat16(*(const v8bf*)(kr1 + 32 + kh), *(const v8bf*)(kr1 + 48 + kh));
#if defined(__gfx1250__)
      s0 = __builtin_amdgcn_wmma_f32_16x16x32_bf16(false, qf0, false, k00, (short)0, s0, false, false);
      s0 = __builtin_amdgcn_wmma_f32_16x16x32_bf16(false, qf1, false, k01, (short)0, s0, false, false);
      s1 = __builtin_amdgcn_wmma_f32_16x16x32_bf16(false, qf0, false, k10, (short)0, s1, false, false);
      s1 = __builtin_amdgcn_wmma_f32_16x16x32_bf16(false, qf1, false, k11, (short)0, s1, false, false);
#endif
    }
    // ---- scale + causal mask (per-lane arithmetic, no branch: EXEC stays full)
#pragma unroll
    for (int v = 0; v < 8; ++v) {
      int rg = qbase + v + 8 * half;
      s0[v] = ((kb + r16)      <= rg) ? s0[v] * SCALE_ : -1e30f;
      s1[v] = ((kb + 16 + r16) <= rg) ? s1[v] * SCALE_ : -1e30f;
    }
    // ---- online softmax (row = v + 8*half; keys live across 16-lane groups)
    float p0[8], p1[8];
#pragma unroll
    for (int v = 0; v < 8; ++v) {
      float cm = fmaxf(s0[v], s1[v]);
      cm = fmaxf(cm, __shfl_xor(cm, 1, 32));
      cm = fmaxf(cm, __shfl_xor(cm, 2, 32));
      cm = fmaxf(cm, __shfl_xor(cm, 4, 32));
      cm = fmaxf(cm, __shfl_xor(cm, 8, 32));
      float nm = fmaxf(m_i[v], cm);
      float f  = __expf(m_i[v] - nm);
      p0[v] = __expf(s0[v] - nm);
      p1[v] = __expf(s1[v] - nm);
      float rs = p0[v] + p1[v];
      rs += __shfl_xor(rs, 1, 32);
      rs += __shfl_xor(rs, 2, 32);
      rs += __shfl_xor(rs, 4, 32);
      rs += __shfl_xor(rs, 8, 32);
      l_i[v] = l_i[v] * f + rs;
      m_i[v] = nm;
#pragma unroll
      for (int j = 0; j < 4; ++j) acc[j][v] = acc[j][v] * f;
    }
    // ---- P -> LDS (score layout -> A-fragment layout via memory)
#pragma unroll
    for (int v = 0; v < 8; ++v) {
      int r = v + 8 * half;
      Ps[wave][r][r16]      = f2bf(p0[v]);
      Ps[wave][r][16 + r16] = f2bf(p1[v]);
    }
    // ---- stage V chunk transposed: Vs[d][key], lane = key
    {
      const __bf16* vrow = V + (((size_t)b * S_ + (kb + lane)) * HKV_ + hk) * HD_;
#pragma unroll
      for (int g = 0; g < 8; ++g) {
        v8bf vv = *(const v8bf*)(vrow + g * 8);
#pragma unroll
        for (int i = 0; i < 8; ++i) Vs[wave][g * 8 + i][lane] = vv[i];
      }
    }
    // wave-private LDS visibility (no block barrier: waves diverge in trip count)
    asm volatile("s_wait_dscnt 0" ::: "memory");
    // ---- O += P V
    v16bf pf = cat16(*(const v8bf*)&Ps[wave][r16][kh], *(const v8bf*)&Ps[wave][r16][16 + kh]);
#pragma unroll
    for (int j = 0; j < 4; ++j) {
      int dn = j * 16 + r16;
      v16bf vf = cat16(*(const v8bf*)&Vs[wave][dn][kh], *(const v8bf*)&Vs[wave][dn][16 + kh]);
#if defined(__gfx1250__)
      acc[j] = __builtin_amdgcn_wmma_f32_16x16x32_bf16(false, pf, false, vf, (short)0, acc[j], false, false);
#endif
    }
    asm volatile("s_wait_dscnt 0" ::: "memory");
  }
  // ---- epilogue: normalize and store [b][s][h][d] bf16 for the Wo GEMM
#pragma unroll
  for (int v = 0; v < 8; ++v) {
    float inv = 1.0f / l_i[v];
    int q = qbase + v + 8 * half;
    __bf16* orow = O + (((size_t)b * S_ + q) * HQ_ + h) * HD_;
#pragma unroll
    for (int j = 0; j < 4; ++j)
      orow[j * 16 + r16] = f2bf(acc[j][v] * inv);
  }
}

// ------------------------------------------------------------------ launch
extern "C" void kernel_launch(void* const* d_in, const int* in_sizes, int n_in,
                              void* d_out, int out_size, void* d_ws, size_t ws_size,
                              hipStream_t stream) {
  (void)in_sizes; (void)n_in; (void)out_size; (void)ws_size;
  const float* x    = (const float*)d_in[0];
  const float* cosb = (const float*)d_in[1];
  const float* sinb = (const float*)d_in[2];
  // d_in[3] = mask: causal mask computed analytically, 1 GB input skipped
  const float* Wq = (const float*)d_in[4];
  const float* Wk = (const float*)d_in[5];
  const float* Wv = (const float*)d_in[6];
  const float* Wo = (const float*)d_in[7];
  float* out = (float*)d_out;

  const size_t M   = (size_t)B_ * S_;        // 4096
  const size_t QN  = (size_t)HQ_ * HD_;      // 2048
  const size_t KVN = (size_t)HKV_ * HD_;     // 512

  char* ws = (char*)d_ws;
  __bf16* Qb = (__bf16*)(ws);                              // 4096x2048
  __bf16* Kb = (__bf16*)(ws + M * QN * 2);                 // 4096x512
  __bf16* Vb = (__bf16*)(ws + M * QN * 2 + M * KVN * 2);
  __bf16* Qr = (__bf16*)(ws + M * QN * 2 + 2 * M * KVN * 2);
  __bf16* Kr = (__bf16*)(ws + 2 * M * QN * 2 + 2 * M * KVN * 2);
  __bf16* AO = (__bf16*)(ws + 2 * M * QN * 2 + 3 * M * KVN * 2);
  // total ws use: ~63 MB

  // QKV projections (fp32 in, bf16 out, bf16 WMMA / fp32 accumulate)
  gemm_wmma<float, __bf16><<<dim3(QN / 128, M / 128), 256, 0, stream>>>(x, Wq, Qb, (int)M, (int)QN, D_);
  gemm_wmma<float, __bf16><<<dim3(KVN / 128, M / 128), 256, 0, stream>>>(x, Wk, Kb, (int)M, (int)KVN, D_);
  gemm_wmma<float, __bf16><<<dim3(KVN / 128, M / 128), 256, 0, stream>>>(x, Wv, Vb, (int)M, (int)KVN, D_);

  // RoPE on Q and K
  {
    int totQ = (int)(M * QN);
    int totK = (int)(M * KVN);
    rope_kernel<<<(totQ + 255) / 256, 256, 0, stream>>>(Qb, cosb, sinb, Qr, HQ_, totQ);
    rope_kernel<<<(totK + 255) / 256, 256, 0, stream>>>(Kb, cosb, sinb, Kr, HKV_, totK);
  }

  // causal flash attention (GQA: 4 Q-heads share each KV head)
  flash_kernel<<<dim3(S_ / 64, B_ * HQ_), 128, 0, stream>>>(Qr, Kr, Vb, AO);

  // output projection (bf16 A staged via async-LDS DMA) -> fp32 d_out
  gemm_wmma<__bf16, float><<<dim3(D_ / 128, M / 128), 256, 0, stream>>>(AO, Wo, out, (int)M, D_, (int)QN);
}
